// MultiHeadAttentionQuantum_65481071396815
// MI455X (gfx1250) — compile-verified, matrix operand order checked by reading, not run on hardware
//
#include <hip/hip_runtime.h>
#include <hip/hip_bf16.h>
#include <math.h>

// MI455X / gfx1250, wave32. D = A(16x32 bf16) x B(32x16 bf16) + C(16x16 f32).
typedef __attribute__((ext_vector_type(16))) __bf16 v16bf;
typedef __attribute__((ext_vector_type(8)))  __bf16 bf16x8;
typedef __attribute__((ext_vector_type(8)))  float  v8f;
typedef __attribute__((ext_vector_type(4)))  float  v4f;
typedef __attribute__((ext_vector_type(4)))  unsigned int v4u;
typedef __attribute__((ext_vector_type(4)))  int    v4i;
typedef __attribute__((ext_vector_type(8)))  int    v8i;

#define B_   2
#define S_   4096
#define E_   1024
#define KT_  32          // keys per attention tile
#define SCALE_ 0.125f    // 1/sqrt(64)

__device__ __forceinline__ v8f wmma_bf16(v16bf a, v16bf b, v8f c) {
  return __builtin_amdgcn_wmma_f32_16x16x32_bf16(
      false, a, false, b, (short)0, c, false, false);
}

// 16-bit fragment, K contiguous in memory (ISA 7.12.2):
// lanes 0-15 hold K = {0..7, 16..23}, lanes 16-31 hold K = {8..15, 24..31},
// VGPR v holds elements 2v,2v+1. Two 16B loads per lane fill the fragment.
__device__ __forceinline__ v16bf frag_kcontig_bf16(const __bf16* base, int hi) {
  v16bf f;
  bf16x8 lo = *(const bf16x8*)(base + hi * 8);
  bf16x8 hh = *(const bf16x8*)(base + 16 + hi * 8);
#pragma unroll
  for (int i = 0; i < 8; ++i) { f[i] = lo[i]; f[8 + i] = hh[i]; }
  return f;
}

__device__ __forceinline__ v16bf frag_kcontig_f32(const float* base, int hi) {
  v16bf f;
  v4f a0 = *(const v4f*)(base + hi * 8);
  v4f a1 = *(const v4f*)(base + hi * 8 + 4);
  v4f a2 = *(const v4f*)(base + 16 + hi * 8);
  v4f a3 = *(const v4f*)(base + 16 + hi * 8 + 4);
#pragma unroll
  for (int i = 0; i < 4; ++i) {
    f[i]      = (__bf16)a0[i];
    f[4 + i]  = (__bf16)a1[i];
    f[8 + i]  = (__bf16)a2[i];
    f[12 + i] = (__bf16)a3[i];
  }
  return f;
}

// Tensor Data Mover: DMA a [KT_ x 1024] bf16 tile (row-major, stride 1024)
// from global into LDS at byte offset lds_off. D# per ISA 08_async_tensor §8.
__device__ __forceinline__ void tdm_load_tile(unsigned lds_off, const __bf16* gsrc) {
  unsigned long long ga = (unsigned long long)(uintptr_t)gsrc;
  v4u g0;
  g0[0] = 1u;                                   // count=1, user descriptor
  g0[1] = lds_off;                              // lds_addr (bytes)
  g0[2] = (unsigned)ga;                         // global_addr[31:0]
  g0[3] = (unsigned)((ga >> 32) & 0x01FFFFFFu)  // global_addr[56:32]
          | 0x80000000u;                        // type = 2 ("image")
  v8i g1;
  g1[0] = 0x00010000;                           // data_size=2B; mask/pad/iter=0
  g1[1] = (int)(1024u << 16);                   // tensor_dim0 = 1024 (lo16)
  g1[2] = (int)(4096u << 16);                   // dim0 hi=0 | tensor_dim1=4096 lo16
  g1[3] = (int)(1024u << 16);                   // dim1 hi=0 | tile_dim0 = 1024
  g1[4] = KT_;                                  // tile_dim1 = 32, tile_dim2 = 0
  g1[5] = 1024;                                 // tensor_dim0_stride lo32
  g1[6] = 0;                                    // stride0 hi | stride1 lo
  g1[7] = 0;
  v4i z4 = {0, 0, 0, 0};
#if defined(__clang_major__) && (__clang_major__ >= 23)
  v8i z8 = {0, 0, 0, 0, 0, 0, 0, 0};
  __builtin_amdgcn_tensor_load_to_lds(g0, g1, z4, z4, z8, 0);
#else
  __builtin_amdgcn_tensor_load_to_lds(g0, g1, z4, z4, 0);
#endif
}

// ---------------------------------------------------------------------------
// Kernel 1: combined[row, h*64+d] = b_dk[d] + sum_q cos(proj[row,h,q]+theta[q])*W_dk[d,q]
// proj[row,h,q] = sum_e x[row,e]*W_proj[h*64+q, e]  (only 128 of 1024 cols used)
// One block = 16 rows; 8 waves, wave w does a 16x16 WMMA tile over K=1024.
// ---------------------------------------------------------------------------
__global__ __launch_bounds__(256) void k_combined(
    const float* __restrict__ x,     // (8192,1024)
    const float* __restrict__ Wp,    // (1024,1024)
    const float* __restrict__ theta, // (8)
    const float* __restrict__ Wdk,   // (64,8)
    const float* __restrict__ bdk,   // (64)
    __bf16* __restrict__ cb)         // (8192,1024) bf16
{
  __shared__ float cosv[16 * 128];

  const int row0 = blockIdx.x * 16;
  const int wave = threadIdx.x >> 5;
  const int lane = threadIdx.x & 31;
  const int hi   = lane >> 4;
  const int ln   = lane & 15;
  const int col0 = wave * 16;

  const int n = col0 + ln;                    // selected col (B frag lane col)
  const int f = ((n >> 3) << 6) | (n & 7);    // W_proj row: h*64 + q

  v8f acc = {0.f, 0.f, 0.f, 0.f, 0.f, 0.f, 0.f, 0.f};

  const float* arow = x  + (size_t)(row0 + ln) * E_;
  const float* brow = Wp + (size_t)f * E_;

  for (int ks = 0; ks < E_; ks += 32) {
    v16bf a = frag_kcontig_f32(arow + ks, hi);
    v16bf b = frag_kcontig_f32(brow + ks, hi);
    acc = wmma_bf16(a, b, acc);
  }

  // C frag element (r, lane): m = r + hi*8, col n. Hardware v_cos_f32 path.
  {
    const float th = theta[n & 7];
#pragma unroll
    for (int r = 0; r < 8; ++r)
      cosv[(r + hi * 8) * 128 + n] = __cosf(acc[r] + th);
  }
  __syncthreads();

  for (int i = threadIdx.x; i < 16 * 1024; i += 256) {
    const int m   = i >> 10;
    const int col = i & 1023;
    const int h   = col >> 6;
    const int d   = col & 63;
    float s = bdk[d];
#pragma unroll
    for (int q = 0; q < 8; ++q)
      s += Wdk[d * 8 + q] * cosv[m * 128 + h * 8 + q];
    cb[(size_t)(row0 + m) * E_ + col] = (__bf16)s;
  }
}

// ---------------------------------------------------------------------------
// Kernel 2: flash attention, Q=K=V=combined (bf16), d = 1024, scale 1/8.
// One block = 16 queries; 8 waves, wave w owns dims [128w, 128w+128).
// K/V tiles staged by TDM into double-buffered dynamic LDS; per 32-key tile:
// partial scores over own dim slice (WMMA), LDS reduce, online softmax,
// rescale O, P@V (WMMA) with V fragments gathered from LDS.
// ---------------------------------------------------------------------------
__global__ __launch_bounds__(256) void k_attn(
    const __bf16* __restrict__ cb,  // (2,4096,1024)
    float* __restrict__ out)        // (2,4096,1024)
{
  extern __shared__ __attribute__((aligned(16))) __bf16 vt[]; // [2][KT_][E_]
  __shared__ float sc[16 * KT_];
  __shared__ float mrow[16], lrow[16], corr[16];

  const int bidx  = blockIdx.x;     // 0..511
  const int batch = bidx >> 8;
  const int q0    = (bidx & 255) * 16;
  const int wave  = threadIdx.x >> 5;
  const int lane  = threadIdx.x & 31;
  const int hi    = lane >> 4;
  const int ln    = lane & 15;
  const int wd0   = wave * 128;

  const __bf16* cbb = cb + (size_t)batch * S_ * E_;
  const unsigned lds_base = (unsigned)(uintptr_t)vt;

  // Preload Q fragments for this wave's 128-dim slice.
  v16bf qf[4];
#pragma unroll
  for (int kk = 0; kk < 4; ++kk)
    qf[kk] = frag_kcontig_bf16(cbb + (size_t)(q0 + ln) * E_ + wd0 + kk * 32, hi);

  v8f o[8];
#pragma unroll
  for (int t = 0; t < 8; ++t)
    o[t] = (v8f){0.f, 0.f, 0.f, 0.f, 0.f, 0.f, 0.f, 0.f};

  if (threadIdx.x < 16) {
    mrow[threadIdx.x] = -__builtin_inff();
    lrow[threadIdx.x] = 0.f;
  }

  // Stage first key/value tile.
  if (wave == 0) tdm_load_tile(lds_base, cbb);

  for (int j0 = 0; j0 < S_; j0 += KT_) {
    const int p = (j0 / KT_) & 1;
    const __bf16* tile = vt + (size_t)p * (KT_ * E_);

    if (wave == 0) __builtin_amdgcn_s_wait_tensorcnt(0);
    __syncthreads();   // A: tile p arrived; previous iteration fully consumed

    // Start DMA of the next tile into the other buffer (overlaps compute).
    if (wave == 0 && j0 + KT_ < S_)
      tdm_load_tile(lds_base + (unsigned)((p ^ 1) * (KT_ * E_ * 2)),
                    cbb + (size_t)(j0 + KT_) * E_);

    sc[threadIdx.x]       = 0.f;
    sc[threadIdx.x + 256] = 0.f;

    // ---- partial scores over this wave's dim slice (K^T from LDS) ----
    v8f c0 = {0.f, 0.f, 0.f, 0.f, 0.f, 0.f, 0.f, 0.f};
    v8f c1 = c0;
#pragma unroll
    for (int kk = 0; kk < 4; ++kk) {
      v16bf b0 = frag_kcontig_bf16(tile + (size_t)ln * E_        + wd0 + kk * 32, hi);
      v16bf b1 = frag_kcontig_bf16(tile + (size_t)(16 + ln) * E_ + wd0 + kk * 32, hi);
      c0 = wmma_bf16(qf[kk], b0, c0);
      c1 = wmma_bf16(qf[kk], b1, c1);
    }
    __syncthreads();   // B: sc zeroed everywhere

    // cross-wave reduction (C frag: m = r + hi*8, n = ln)
#pragma unroll
    for (int r = 0; r < 8; ++r) {
      const int m = r + hi * 8;
      atomicAdd(&sc[m * KT_ + ln],      c0[r]);
      atomicAdd(&sc[m * KT_ + 16 + ln], c1[r]);
    }
    __syncthreads();   // C

    // online softmax update (16 lanes, one per query row)
    if (threadIdx.x < 16) {
      const int m = threadIdx.x;
      const float mo = mrow[m];
      float mx = mo;
      for (int jj = 0; jj < KT_; ++jj)
        mx = fmaxf(mx, SCALE_ * sc[m * KT_ + jj]);
      const float cfac = __expf(mo - mx);   // exp(-inf)=0 on first tile
      float sum = 0.f;
      for (int jj = 0; jj < KT_; ++jj) {
        const float pv = __expf(SCALE_ * sc[m * KT_ + jj] - mx);
        sc[m * KT_ + jj] = pv;
        sum += pv;
      }
      mrow[m] = mx;
      lrow[m] = lrow[m] * cfac + sum;
      corr[m] = cfac;
    }
    __syncthreads();   // D

    // P fragment (A: m = ln, k = key; contiguous in sc row) fp32 -> bf16
    v16bf pf;
    {
      const float* pr = &sc[ln * KT_];
      v4f p0 = *(const v4f*)(pr + hi * 8);
      v4f p1 = *(const v4f*)(pr + hi * 8 + 4);
      v4f p2 = *(const v4f*)(pr + 16 + hi * 8);
      v4f p3 = *(const v4f*)(pr + 16 + hi * 8 + 4);
#pragma unroll
      for (int i = 0; i < 4; ++i) {
        pf[i]      = (__bf16)p0[i];
        pf[4 + i]  = (__bf16)p1[i];
        pf[8 + i]  = (__bf16)p2[i];
        pf[12 + i] = (__bf16)p3[i];
      }
    }

    // rescale O accumulators
    float cr[8];
#pragma unroll
    for (int r = 0; r < 8; ++r) cr[r] = corr[r + hi * 8];
#pragma unroll
    for (int t = 0; t < 8; ++t)
#pragma unroll
      for (int r = 0; r < 8; ++r) o[t][r] *= cr[r];

    // O += P(16x32) @ V(32 x 128-slice); V frag gathered from LDS tile
#pragma unroll
    for (int t = 0; t < 8; ++t) {
      const int dim = wd0 + t * 16 + ln;
      v16bf bv;
#pragma unroll
      for (int v = 0; v < 8; ++v) {
        const int kb = ((v < 4) ? (2 * v) : (8 + 2 * v)) + hi * 8;
        bv[2 * v]     = tile[(size_t)kb * E_ + dim];
        bv[2 * v + 1] = tile[(size_t)(kb + 1) * E_ + dim];
      }
      o[t] = wmma_bf16(pf, bv, o[t]);
    }
  }

  // finalize: divide by softmax denominator, store fp32
  float li[8];
#pragma unroll
  for (int r = 0; r < 8; ++r) li[r] = 1.0f / lrow[r + hi * 8];

  float* ob = out + (size_t)batch * S_ * E_;
#pragma unroll
  for (int t = 0; t < 8; ++t)
#pragma unroll
    for (int r = 0; r < 8; ++r)
      ob[(size_t)(q0 + r + hi * 8) * E_ + wd0 + t * 16 + ln] = o[t][r] * li[r];
}

extern "C" void kernel_launch(void* const* d_in, const int* in_sizes, int n_in,
                              void* d_out, int out_size, void* d_ws, size_t ws_size,
                              hipStream_t stream) {
  (void)in_sizes; (void)n_in; (void)out_size; (void)ws_size;
  const float* x     = (const float*)d_in[0];
  const float* Wp    = (const float*)d_in[1];
  const float* theta = (const float*)d_in[2];
  const float* Wdk   = (const float*)d_in[3];
  const float* bdk   = (const float*)d_in[4];
  float* out = (float*)d_out;
  __bf16* cb = (__bf16*)d_ws;   // 8192*1024 bf16 = 16 MB scratch (L2-resident)

  const int dynLds = 2 * KT_ * E_ * (int)sizeof(__bf16);  // 128 KB (<320 KB/WGP)
  hipFuncSetAttribute(reinterpret_cast<const void*>(k_attn),
                      hipFuncAttributeMaxDynamicSharedMemorySize, dynLds);

  k_combined<<<dim3((B_ * S_) / 16), dim3(256), 0, stream>>>(x, Wp, theta, Wdk, bdk, cb);
  k_attn<<<dim3(B_ * (S_ / 16)), dim3(256), dynLds, stream>>>(cb, out);
}
